// AttentionalAggregator_70841190580839
// MI455X (gfx1250) — matrix operand order repaired and checked by reading, not compiled, not run on hardware
//
#include <hip/hip_runtime.h>

typedef __attribute__((ext_vector_type(16))) __bf16 v16bf;
typedef __attribute__((ext_vector_type(8)))  float  v8f;

union BF16x16 { unsigned int u[8]; v16bf v; };

// fp32 x2 -> packed bf16 pair (lo in [15:0], hi in [31:16]).
#if __has_builtin(__builtin_amdgcn_cvt_pk_bf16_f32)
typedef __attribute__((ext_vector_type(2))) __bf16 v2bf;
__device__ __forceinline__ unsigned int pack_bf2(float lo, float hi) {
  v2bf r = __builtin_amdgcn_cvt_pk_bf16_f32(lo, hi);
  return __builtin_bit_cast(unsigned int, r);
}
#else
// Round-half-up then byte-select the two high halves with one v_perm_b32.
__device__ __forceinline__ unsigned int pack_bf2(float lo, float hi) {
  unsigned int ulo = __builtin_bit_cast(unsigned int, lo) + 0x8000u;
  unsigned int uhi = __builtin_bit_cast(unsigned int, hi) + 0x8000u;
  // D.byte0 = lo.byte2, D.byte1 = lo.byte3, D.byte2 = hi.byte2, D.byte3 = hi.byte3
  return __builtin_amdgcn_perm(uhi, ulo, 0x07060302u);
}
#endif

// F = D = 128, N = 25, rows/(b,h) = 26.
// 4 groups per block -> 104 rows, padded to 112 (7 WMMA M-tiles).
#define GPB        4
#define ROWS       104           // GPB * 26
#define MPAD       112           // 7 * 16
#define MT         7
#define XPK_STRIDE 65            // u32 pairs/row (64 used) + 1 -> conflict-free A reads
#define XW_STRIDE  132           // fp32/row (128 used) + 4    -> conflict-free C stores

__global__ __launch_bounds__(256)
void gat_aggr_kernel(const float* __restrict__ x_self,
                     const float* __restrict__ x_neigh,
                     const float* __restrict__ w,
                     const float* __restrict__ a_self,
                     const float* __restrict__ a_neigh,
                     float* __restrict__ out, int BH) {
  __shared__ unsigned int s_xpk[MPAD * XPK_STRIDE];   // bf16-pair A tiles (29,120 B)
  __shared__ float        s_xw[MPAD * XW_STRIDE];     // fp32 GEMM result  (59,136 B)
  __shared__ float        s_sn[GPB][32];
  __shared__ float        s_ss[GPB];
  __shared__ float        s_attn[GPB][32];

  const int tid  = threadIdx.x;
  const int lane = tid & 31;
  const int wv   = tid >> 5;        // 8 waves
  const int bh0  = blockIdx.x * GPB;

  // ---- Stage x rows as packed bf16 pairs; rows >= ROWS are zero padding.
  //      64 consecutive threads cover one 128-float row -> coalesced float2 loads.
  for (int i = tid; i < MPAD * 64; i += 256) {
    int r = i >> 6, p = i & 63;
    float lo = 0.f, hi = 0.f;
    if (r < ROWS) {
      int g = r / 26, j = r - g * 26;
      int bh = bh0 + g;
      if (bh < BH) {
        const float* src = (j == 0) ? (x_self + (size_t)bh * 128)
                                    : (x_neigh + ((size_t)bh * 25 + (j - 1)) * 128);
        const float2 v2 = ((const float2*)src)[p];
        lo = v2.x; hi = v2.y;
      }
    }
    s_xpk[r * XPK_STRIDE + p] = pack_bf2(lo, hi);
  }

  // ---- Each wave loads its B fragments (w columns n0..n0+15) from global once.
  //      B k order chosen here (contiguous, same as A): VGPR j holds K = {2j, 2j+1}; lanes 0-15 -> K 0-15,
  //      lanes 16-31 -> K 16-31; lane%16 = column n.
  const int n0 = wv * 16;
  const int kh = lane >> 4;
  const int nl = lane & 15;
  BF16x16 Bf[4];
#pragma unroll
  for (int kk = 0; kk < 4; ++kk) {
#pragma unroll
    for (int j = 0; j < 8; ++j) {
      int k2 = kk * 16 + kh * 8 + j;                     // pair index: K = 2*k2, 2*k2+1
      float lo = w[(size_t)(2 * k2)     * 128 + n0 + nl];
      float hi = w[(size_t)(2 * k2 + 1) * 128 + n0 + nl];
      Bf[kk].u[j] = pack_bf2(lo, hi);
    }
  }

  __syncthreads();

  // ---- GEMM: [112 x 128] x [128 x 128]; wave wv owns output columns n0..n0+15.
  //      A fragments double-buffered so DS loads of tile mt+1 overlap the WMMAs of
  //      tile mt (hides LDS latency, avoids WMMA->VALU WAR hazard stalls).
  //      A k order chosen here (contiguous, same as B): lane%16 = row M; lanes 0-15: v0-7 K0-15;
  //      lanes 16-31: v0-7 K16-31.
  {
    BF16x16 Af[2][4];
    auto loadA = [&](BF16x16 (&A)[4], int mt) {
      const int rb = (mt * 16 + nl) * XPK_STRIDE;
#pragma unroll
      for (int kk = 0; kk < 4; ++kk)
#pragma unroll
        for (int j = 0; j < 8; ++j) {
          int k2p = kh * 8 + j;
          A[kk].u[j] = s_xpk[rb + kk * 16 + k2p];
        }
    };

    loadA(Af[0], 0);
#pragma unroll
    for (int mt = 0; mt < MT; ++mt) {
      const int cur = mt & 1;
      if (mt < MT - 1) loadA(Af[cur ^ 1], mt + 1);   // prefetch next M-tile
      v8f acc = {};
#pragma unroll
      for (int kk = 0; kk < 4; ++kk)
        acc = __builtin_amdgcn_wmma_f32_16x16x32_bf16(false, Af[cur][kk].v,
                                                      false, Bf[kk].v,
                                                      (short)0, acc, false, false);
      // C layout: VGPR r -> row M = r + 8*(lane/16), column = n0 + lane%16.
#pragma unroll
      for (int r = 0; r < 8; ++r)
        s_xw[(mt * 16 + r + 8 * kh) * XW_STRIDE + n0 + nl] = acc[r];
    }
  }

  __syncthreads();

  // ---- Attention scores (fp32): per group, dot each of 26 rows with a_neigh,
  //      and row 0 with a_self. 2 waves per group, lane covers 4 columns.
  {
    int g = wv >> 1, sub = wv & 1;
    float an[4], as[4];
#pragma unroll
    for (int c = 0; c < 4; ++c) { an[c] = a_neigh[lane * 4 + c]; as[c] = a_self[lane * 4 + c]; }
    for (int j = sub; j < 26; j += 2) {
      const float* row = &s_xw[(g * 26 + j) * XW_STRIDE];
      float dn = 0.f, dsf = 0.f;
#pragma unroll
      for (int c = 0; c < 4; ++c) {
        float v = row[lane * 4 + c];
        dn += v * an[c];
        if (j == 0) dsf += v * as[c];
      }
#pragma unroll
      for (int off = 16; off > 0; off >>= 1) {
        dn += __shfl_xor(dn, off, 32);
        if (j == 0) dsf += __shfl_xor(dsf, off, 32);
      }
      if (lane == 0) {
        s_sn[g][j] = dn;
        if (j == 0) s_ss[g] = dsf;
      }
    }
  }

  __syncthreads();

  // ---- leaky_relu(0.2) + stable softmax over the 26 entries (wave g -> group g).
  if (wv < GPB) {
    int g = wv;
    float u = -3.4e38f;
    if (lane < 26) {
      float t = s_ss[g] + s_sn[g][lane];
      u = (t > 0.f) ? t : 0.2f * t;
    }
    float mx = u;
#pragma unroll
    for (int off = 16; off > 0; off >>= 1) {
      float o = __shfl_xor(mx, off, 32);
      mx = (o > mx) ? o : mx;
    }
    float e = (lane < 26) ? __expf(u - mx) : 0.f;
    float sum = e;
#pragma unroll
    for (int off = 16; off > 0; off >>= 1) sum += __shfl_xor(sum, off, 32);
    if (lane < 26) s_attn[g][lane] = e / sum;
  }

  __syncthreads();

  // ---- Weighted sum over the 26 rows + ReLU; coalesced store.
  for (int i = tid; i < GPB * 128; i += 256) {
    int g = i >> 7, n = i & 127;
    int bh = bh0 + g;
    if (bh < BH) {
      float acc = 0.f;
#pragma unroll
      for (int k = 0; k < 26; ++k)
        acc += s_attn[g][k] * s_xw[(g * 26 + k) * XW_STRIDE + n];
      out[(size_t)bh * 128 + n] = acc > 0.f ? acc : 0.f;
    }
  }
}

extern "C" void kernel_launch(void* const* d_in, const int* in_sizes, int n_in,
                              void* d_out, int out_size, void* d_ws, size_t ws_size,
                              hipStream_t stream) {
  const float* x_self  = (const float*)d_in[0];
  const float* x_neigh = (const float*)d_in[1];
  const float* w_feat  = (const float*)d_in[2];
  const float* a_self  = (const float*)d_in[3];
  const float* a_neigh = (const float*)d_in[4];
  float* out = (float*)d_out;

  const int BH = in_sizes[0] / 128;            // B*H (F = 128)
  const int grid = (BH + GPB - 1) / GPB;       // 4 (b,h) groups per block

  hipLaunchKernelGGL(gat_aggr_kernel, dim3(grid), dim3(256), 0, stream,
                     x_self, x_neigh, w_feat, a_self, a_neigh, out, BH);
}
